// Channel_map_80951543595345
// MI455X (gfx1250) — matrix operand order repaired and checked by reading, not compiled, not run on hardware
//
#include <hip/hip_runtime.h>

// Problem constants
constexpr int Bc   = 16;     // batch
constexpr int Tc   = 5;
constexpr int Cc   = 512;
constexpr int HWc  = 256;    // 16*16
constexpr int KIN  = 1280;   // T*H*W
constexpr int NF   = 10;
constexpr int GOUT = 2560;   // NF*H*W

typedef __attribute__((ext_vector_type(16))) __bf16 v16bf;
typedef __attribute__((ext_vector_type(8)))  float  v8f;
typedef int v4i __attribute__((vector_size(16)));   // matches builtin param type

#define AS1 __attribute__((address_space(1)))
#define AS3 __attribute__((address_space(3)))

#if defined(__has_builtin)
#if __has_builtin(__builtin_amdgcn_global_load_async_to_lds_b128) && \
    __has_builtin(__builtin_amdgcn_s_wait_asynccnt)
#define HAVE_ASYNC_LDS 1
#endif
#endif

// 16-byte global -> LDS copy; async (ASYNCcnt) when available.
__device__ __forceinline__ void async_cp16(const __bf16* g, __bf16* l) {
#ifdef HAVE_ASYNC_LDS
    __builtin_amdgcn_global_load_async_to_lds_b128(
        (AS1 v4i*)g, (AS3 v4i*)l, 0, 0);
#else
    *(uint4*)l = *(const uint4*)g;
#endif
}

__device__ __forceinline__ void async_wait_all() {
#ifdef HAVE_ASYNC_LDS
    __builtin_amdgcn_s_wait_asynccnt(0);
#endif
}

// ---------------------------------------------------------------------------
// Kernel 1: convert G3_w -> bf16, compute wmean = mean_f conv1_w, bmean
// ---------------------------------------------------------------------------
__global__ void prep_kernel(const float* __restrict__ G3_w,
                            const float* __restrict__ conv1_w,
                            const float* __restrict__ conv1_b,
                            __bf16* __restrict__ wA,
                            float* __restrict__ wmean,
                            float* __restrict__ bmean) {
    long gid = (long)blockIdx.x * blockDim.x + threadIdx.x;
    const long n = (long)GOUT * KIN;
    if (gid < n) wA[gid] = (__bf16)G3_w[gid];
    if (gid < KIN) {
        float s = 0.f;
        #pragma unroll
        for (int f = 0; f < NF; ++f) s += conv1_w[f * KIN + gid];
        wmean[gid] = s * (1.0f / NF);
    }
    if (gid == 0) {
        float s = 0.f;
        #pragma unroll
        for (int f = 0; f < NF; ++f) s += conv1_b[f];
        *bmean = s * (1.0f / NF);
    }
}

// ---------------------------------------------------------------------------
// Kernel 2: input [B,T,C,H,W] -> xT bf16 [B][C][K] (K contiguous),
//           fused pooled[b,c] = sum_k x*wmean + bmean   (conv1 + avgpool)
// ---------------------------------------------------------------------------
__global__ void xform_kernel(const float* __restrict__ input,
                             const float* __restrict__ wmean,
                             const float* __restrict__ bmean,
                             __bf16* __restrict__ xT,
                             float* __restrict__ pooled) {
    const int b = blockIdx.y, c = blockIdx.x, tid = threadIdx.x; // 256 threads = hw
    float part = 0.f;
    const long obase = ((long)(b * Cc + c)) * KIN;
    #pragma unroll
    for (int t = 0; t < Tc; ++t) {
        long ii = ((long)((b * Tc + t) * Cc + c)) * HWc + tid;
        float v = input[ii];
        int k = t * HWc + tid;
        xT[obase + k] = (__bf16)v;
        part += v * wmean[k];
    }
    __shared__ float red[256];
    red[tid] = part;
    __syncthreads();
    for (int s = 128; s > 0; s >>= 1) {
        if (tid < s) red[tid] += red[tid + s];
        __syncthreads();
    }
    if (tid == 0) pooled[b * Cc + c] = red[0] + *bmean;
}

// ---------------------------------------------------------------------------
// Kernel 3: pre = pooled @ ffnn1^T + b          (tiny: 16x512x512)
// ---------------------------------------------------------------------------
__global__ void pre_kernel(const float* __restrict__ pooled,
                           const float* __restrict__ ffnn1_w,
                           const float* __restrict__ ffnn1_b,
                           float* __restrict__ pre) {
    const int b = blockIdx.x, c = threadIdx.x; // 512 threads
    __shared__ float sp[Cc];
    sp[c] = pooled[b * Cc + c];
    __syncthreads();
    float acc = 0.f;
    const float* wr = ffnn1_w + (long)c * Cc;
    for (int j = 0; j < Cc; ++j) acc += sp[j] * wr[j];
    pre[b * Cc + c] = acc + ffnn1_b[c];
}

// ---------------------------------------------------------------------------
// Kernel 4: scale = a0*relu + a1*sigmoid + a2*softmax_C(pre)
// ---------------------------------------------------------------------------
__global__ void scale_kernel(const float* __restrict__ pre,
                             const float* __restrict__ aw,
                             float* __restrict__ scl) {
    const int b = blockIdx.x, c = threadIdx.x; // 512 threads
    __shared__ float red[Cc];
    float p = pre[b * Cc + c];
    red[c] = p;
    __syncthreads();
    for (int s = 256; s > 0; s >>= 1) {
        if (c < s) red[c] = fmaxf(red[c], red[c + s]);
        __syncthreads();
    }
    float m = red[0];
    __syncthreads();
    float e = __expf(p - m);
    red[c] = e;
    __syncthreads();
    for (int s = 256; s > 0; s >>= 1) {
        if (c < s) red[c] += red[c + s];
        __syncthreads();
    }
    float sum = red[0];
    float a0 = aw[0], a1 = aw[1], a2 = aw[2];
    scl[b * Cc + c] = a0 * fmaxf(p, 0.f)
                    + a1 * (1.f / (1.f + __expf(-p)))
                    + a2 * (e / sum);
}

// ---------------------------------------------------------------------------
// Kernel 5: big GEMM  D[g,c] = sum_k wA[g,k]*xT[b,c,k], fused epilogue
//   out[b, g>>8, c, g&255] = scale[b,c] * (D + G3_b[g])
// 256 threads = 8 waves; block tile 128(M=g) x 128(N=c), BK=32
// wave tile 64x32 = 4x2 wmma_f32_16x16x32_bf16 accumulators
// Double-buffered LDS fed by async global->LDS copies (ASYNCcnt).
// ---------------------------------------------------------------------------
#define LDST 56   // LDS row stride in bf16 (112 B = 7*16 B): m*28 mod 64 hits
                  // 16 distinct bank quads -> conflict-free b128 frag reads
constexpr int KT = KIN / 32;  // 40 K-tiles

__global__ __launch_bounds__(256) void gemm_kernel(
        const __bf16* __restrict__ wA,
        const __bf16* __restrict__ xT,
        const float*  __restrict__ G3_b,
        const float*  __restrict__ scl,
        float*        __restrict__ out) {
    // 4 x 14336 B = 57344 B static LDS
    __shared__ __align__(16) __bf16 As[2][128 * LDST];
    __shared__ __align__(16) __bf16 Bs[2][128 * LDST];

    const int tid  = threadIdx.x;
    const int lane = tid & 31, wave = tid >> 5;
    const int half = lane >> 4, l16 = lane & 15;
    const int wm = wave & 1;   // M half of block tile (0/1 -> 64 rows)
    const int wn = wave >> 1;  // N quarter (0..3 -> 32 cols)
    const int b     = blockIdx.z;
    const int gBase = blockIdx.y * 128;
    const int cBase = blockIdx.x * 128;

    const __bf16* Ag = wA + (long)gBase * KIN;
    const __bf16* Bg = xT + ((long)b * Cc + cBase) * KIN;

    // This thread's two 16B chunks of each 128x32 tile
    const int ci0 = tid, ci1 = tid + 256;
    const int row0 = ci0 >> 2, c80 = (ci0 & 3) * 8;
    const int row1 = ci1 >> 2, c81 = (ci1 & 3) * 8;

    v8f acc[4][2] = {};

    // Prologue: fill buffer 0
    {
        async_cp16(&Ag[(long)row0 * KIN + c80], &As[0][row0 * LDST + c80]);
        async_cp16(&Bg[(long)row0 * KIN + c80], &Bs[0][row0 * LDST + c80]);
        async_cp16(&Ag[(long)row1 * KIN + c81], &As[0][row1 * LDST + c81]);
        async_cp16(&Bg[(long)row1 * KIN + c81], &Bs[0][row1 * LDST + c81]);
        async_wait_all();
        __syncthreads();
    }

    for (int kt = 0; kt < KT; ++kt) {
        const int cur = kt & 1, nxt = cur ^ 1;
        // Issue async copies for the next K-slab into the other buffer
        if (kt + 1 < KT) {
            const int k0 = (kt + 1) * 32;
            async_cp16(&Ag[(long)row0 * KIN + k0 + c80], &As[nxt][row0 * LDST + c80]);
            async_cp16(&Bg[(long)row0 * KIN + k0 + c80], &Bs[nxt][row0 * LDST + c80]);
            async_cp16(&Ag[(long)row1 * KIN + k0 + c81], &As[nxt][row1 * LDST + c81]);
            async_cp16(&Bg[(long)row1 * KIN + k0 + c81], &Bs[nxt][row1 * LDST + c81]);
        }

        // Compute on the current buffer
        union Frag { v16bf v; uint4 q[2]; };
        Frag af[4], bfr[2];
        // A frag: lane holds row m; elem e: K = 16*(e/8) + 8*half + e%8
        #pragma unroll
        for (int mt = 0; mt < 4; ++mt) {
            int m = wm * 64 + mt * 16 + l16;
            const __bf16* p = &As[cur][m * LDST];
            af[mt].q[0] = *(const uint4*)&p[half * 8];
            af[mt].q[1] = *(const uint4*)&p[16 + half * 8];
        }
        // B frag: lane holds col n; elem e: K = 16*(e/8) + 8*half + e%8
        #pragma unroll
        for (int nt = 0; nt < 2; ++nt) {
            int n = wn * 32 + nt * 16 + l16;
            const __bf16* p = &Bs[cur][n * LDST];
            bfr[nt].q[0] = *(const uint4*)&p[half * 8];
            bfr[nt].q[1] = *(const uint4*)&p[16 + half * 8];
        }
        #pragma unroll
        for (int mt = 0; mt < 4; ++mt)
            #pragma unroll
            for (int nt = 0; nt < 2; ++nt)
                acc[mt][nt] = __builtin_amdgcn_wmma_f32_16x16x32_bf16(
                    false, af[mt].v, false, bfr[nt].v,
                    (short)0, acc[mt][nt], false, false);

        // Own async copies done, then workgroup-wide rendezvous before swap
        async_wait_all();
        __syncthreads();
    }

    // Epilogue: C/D layout: lane n = l16, elem r -> m = 8*half + r.
    // 8 consecutive g for fixed c -> 8 contiguous output floats.
    #pragma unroll
    for (int nt = 0; nt < 2; ++nt) {
        int c = cBase + wn * 32 + nt * 16 + l16;
        float s = scl[b * Cc + c];
        #pragma unroll
        for (int mt = 0; mt < 4; ++mt) {
            int g0 = gBase + wm * 64 + mt * 16 + half * 8;
            int f  = g0 >> 8, hw = g0 & 255;
            float* op = out + (((long)(b * NF + f) * Cc + c) * HWc + hw);
            float t0[8];
            #pragma unroll
            for (int r = 0; r < 8; ++r)
                t0[r] = (acc[mt][nt][r] + G3_b[g0 + r]) * s;
            float4 v0 = {t0[0], t0[1], t0[2], t0[3]};
            float4 v1 = {t0[4], t0[5], t0[6], t0[7]};
            *(float4*)op       = v0;
            *(float4*)(op + 4) = v1;
        }
    }
}

// ---------------------------------------------------------------------------
extern "C" void kernel_launch(void* const* d_in, const int* in_sizes, int n_in,
                              void* d_out, int out_size, void* d_ws, size_t ws_size,
                              hipStream_t stream) {
    const float* input   = (const float*)d_in[0];
    // d_in[1] (input1) is unused by the reference
    const float* act_w   = (const float*)d_in[2];
    const float* conv1_w = (const float*)d_in[3];
    const float* conv1_b = (const float*)d_in[4];
    const float* G3_w    = (const float*)d_in[5];
    const float* G3_b    = (const float*)d_in[6];
    const float* ffnn1_w = (const float*)d_in[7];
    const float* ffnn1_b = (const float*)d_in[8];
    float* out = (float*)d_out;

    // Workspace layout (all 16B aligned): ~27.6 MB total
    char* ws = (char*)d_ws;
    __bf16* wA     = (__bf16*)ws;                         //  6,553,600 B
    __bf16* xT     = (__bf16*)(ws + 6553600);             // 20,971,520 B
    float*  pooled = (float*)(ws + 27525120);             //     32,768 B
    float*  pre    = pooled + Bc * Cc;
    float*  scl    = pre    + Bc * Cc;
    float*  wmean  = scl    + Bc * Cc;
    float*  bmean  = wmean  + KIN;

    prep_kernel<<<(GOUT * KIN + 255) / 256, 256, 0, stream>>>(
        G3_w, conv1_w, conv1_b, wA, wmean, bmean);
    xform_kernel<<<dim3(Cc, Bc), 256, 0, stream>>>(
        input, wmean, bmean, xT, pooled);
    pre_kernel<<<Bc, Cc, 0, stream>>>(pooled, ffnn1_w, ffnn1_b, pre);
    scale_kernel<<<Bc, Cc, 0, stream>>>(pre, act_w, scl);
    gemm_kernel<<<dim3(Cc / 128, GOUT / 128, Bc), 256, 0, stream>>>(
        wA, xT, G3_b, scl, out);
}
